// DLinear_45990509805636
// MI455X (gfx1250) — compile-verified, matrix operand order checked by reading, not run on hardware
//
#include <hip/hip_runtime.h>

// ---------------------------------------------------------------------------
// DLinear fused kernel for MI455X (gfx1250, wave32, WMMA + TDM).
//   trend[b,t,c] = (S[b,c] + (18-t)*x[b,0,c] + (t+2)*x[b,16,c]) / 37
//   out[b,t,d]   = x[b,t,:]·Ws[t,d,:] + trend[b,t,:]·(Wt-Ws)[t,d,:] + bs+bt
// bf16 inputs, f32 WMMA accumulation; GEMM staging via Tensor Data Mover
// (tensor_load_to_lds) with double-buffered LDS, TENSORcnt pipelining.
// ---------------------------------------------------------------------------

typedef __attribute__((ext_vector_type(16))) __bf16 v16bf;
typedef __attribute__((ext_vector_type(8)))  float  v8f;
typedef __attribute__((ext_vector_type(4)))  float  f32x4;
typedef __attribute__((ext_vector_type(4)))  __bf16 bf16x4;
typedef __attribute__((ext_vector_type(4)))  unsigned int u32x4;
typedef __attribute__((ext_vector_type(4)))  int          i32x4;
typedef __attribute__((ext_vector_type(8)))  int          i32x8;

constexpr int T  = 17;
constexpr int C  = 512;   // Cin
constexpr int D  = 512;   // Cout
constexpr int KS = 37;    // moving-average window

// GEMM tiling
constexpr int BM = 128;
constexpr int BN = 128;
constexpr int BK = 32;
constexpr int LDS_STRIDE = BK + 8;            // 40 bf16 = 80B row (64B data + 16B pad)
constexpr int TILE_ELEMS = BM * LDS_STRIDE;   // one 128x32 tile (padded) in bf16 elems
constexpr int BUF_ELEMS  = 4 * TILE_ELEMS;    // Ax, At, Bs, Bd
constexpr int NK = C / BK;                    // 16 K-steps

// ---------------------------------------------------------------------------
// Phase 1: x fp32 -> xh bf16 [t][b][c], trend bf16 [t][b][c]
// ---------------------------------------------------------------------------
__global__ void prep_x_kernel(const float* __restrict__ x,
                              __bf16* __restrict__ xh,
                              __bf16* __restrict__ th, int B)
{
    const int nC4 = C / 4;
    int idx = blockIdx.x * blockDim.x + threadIdx.x;   // b*(C/4)+c4
    if (idx >= B * nC4) return;
    int b  = idx / nC4;
    int c4 = idx - b * nC4;

    const f32x4* xp = reinterpret_cast<const f32x4*>(x) + (size_t)b * T * nC4 + c4;
    f32x4 v[T];
    f32x4 S = {0.f, 0.f, 0.f, 0.f};
#pragma unroll
    for (int t = 0; t < T; ++t) { v[t] = xp[(size_t)t * nC4]; S += v[t]; }

    const float inv = 1.0f / (float)KS;
#pragma unroll
    for (int t = 0; t < T; ++t) {
        f32x4 tr = (S + (float)(18 - t) * v[0] + (float)(t + 2) * v[T - 1]) * inv;
        size_t o = ((size_t)t * B + b) * (size_t)nC4 + c4;
        reinterpret_cast<bf16x4*>(xh)[o] = __builtin_convertvector(v[t], bf16x4);
        reinterpret_cast<bf16x4*>(th)[o] = __builtin_convertvector(tr,   bf16x4);
    }
}

// ---------------------------------------------------------------------------
// Phase 2: Ws -> bf16, (Wt - Ws) -> bf16   (layout [t][d][c], same as input)
// ---------------------------------------------------------------------------
__global__ void prep_w_kernel(const float* __restrict__ Ws,
                              const float* __restrict__ Wt,
                              __bf16* __restrict__ wsh,
                              __bf16* __restrict__ wdh, int n4)
{
    int i = blockIdx.x * blockDim.x + threadIdx.x;
    if (i >= n4) return;
    f32x4 s = reinterpret_cast<const f32x4*>(Ws)[i];
    f32x4 w = reinterpret_cast<const f32x4*>(Wt)[i];
    reinterpret_cast<bf16x4*>(wsh)[i] = __builtin_convertvector(s, bf16x4);
    reinterpret_cast<bf16x4*>(wdh)[i] = __builtin_convertvector(w - s, bf16x4);
}

// ---------------------------------------------------------------------------
// TDM: issue one 64-row x 64-byte subtile load (global -> LDS) with row
// padding 64B data + 16B pad, i.e. LDS row stride = 80B = LDS_STRIDE bf16.
// D# per CDNA5 ISA. This toolchain's builtin is the
// 6-arg form: (group0, group1, group2, group3, extra, cpol) — unused groups 0.
// ---------------------------------------------------------------------------
#if __has_builtin(__builtin_amdgcn_tensor_load_to_lds)
#define USE_TDM 1
__device__ __forceinline__ void tdm_load_subtile(unsigned lds_addr,
                                                 unsigned long long gaddr)
{
    // group0: count=1 | lds_addr | global_addr | type=2
    u32x4 g0 = { 1u,
                 lds_addr,
                 (unsigned)(gaddr & 0xFFFFFFFFull),
                 (unsigned)((gaddr >> 32) & 0x01FFFFFFull) | (2u << 30) };
    // group1: data_size=3(8B), pad_enable, pad_interval=3(64B), pad_amount=3(16B)
    //         tensor_dim0=128 units(1KB row), tensor_dim1=1<<20,
    //         tile_dim0=8 units(64B), tile_dim1=64 rows, dim0_stride=128 units
    const unsigned td0 = 128u, td1 = (1u << 20), tile0 = 8u, tile1 = 64u, str0 = 128u;
    i32x8 g1 = { (int)((3u << 16) | (1u << 20) | (3u << 22) | (3u << 25)),
                 (int)((td0 & 0xFFFFu) << 16),
                 (int)(((td0 >> 16) & 0xFFFFu) | ((td1 & 0xFFFFu) << 16)),
                 (int)(((td1 >> 16) & 0xFFFFu) | (tile0 << 16)),
                 (int)tile1,
                 (int)str0,
                 0, 0 };
    i32x4 z4 = { 0, 0, 0, 0 };
    i32x8 z8 = { 0, 0, 0, 0, 0, 0, 0, 0 };
    __builtin_amdgcn_tensor_load_to_lds(g0, g1, z4, z4, z8, 0);
}
#else
#define USE_TDM 0
#endif

// ---------------------------------------------------------------------------
// Phase 3: per-token fused dual GEMM with shared f32 accumulators.
// Block: 256 threads = 8 waves (4 along M x 2 along N), tile 128x128, K-step 32.
// Wave tile: 32x64 -> 2x4 fragments of 16x16; 16 v_wmma per wave per K-step.
// Staging: 8 waves each TDM-load one 64-row subtile into the next LDS buffer.
// ---------------------------------------------------------------------------
__global__ __launch_bounds__(256)
void dlinear_gemm_kernel(const __bf16* __restrict__ xh,
                         const __bf16* __restrict__ th,
                         const __bf16* __restrict__ wsh,
                         const __bf16* __restrict__ wdh,
                         const float*  __restrict__ bsv,
                         const float*  __restrict__ btv,
                         float* __restrict__ out, int B)
{
    __shared__ __bf16 smem[2 * BUF_ELEMS];   // 2 x 40KB double buffer

    const int t   = blockIdx.z;
    const int mb  = blockIdx.x * BM;
    const int nb  = blockIdx.y * BN;
    const int tid = threadIdx.x;

    const int wave = tid >> 5;
    const int lane = tid & 31;
    const int lo   = lane & 15;
    const int hi   = lane >> 4;
    const int wm   = (wave & 3) * 32;   // wave M offset in tile
    const int wn   = (wave >> 2) * 64;  // wave N offset in tile

#if USE_TDM
    // ---- per-wave subtile descriptor base (uniform via readfirstlane) ----
    const int w  = __builtin_amdgcn_readfirstlane((int)(threadIdx.x >> 5));
    const int tw = w >> 1;           // 0:Ax(x) 1:At(trend) 2:Bs(Ws) 3:Bd(Wt-Ws)
    const int hh = w & 1;            // subtile half: rows [hh*64, hh*64+64)
    const __bf16* srcb = (tw == 0) ? xh : (tw == 1) ? th : (tw == 2) ? wsh : wdh;
    const size_t rowbase = (tw < 2) ? ((size_t)t * B + mb) : ((size_t)t * D + nb);
    const unsigned long long gsub =
        (unsigned long long)(size_t)(srcb + (rowbase + (size_t)hh * 64) * C);
    const unsigned lds_base = (unsigned)(size_t)(void*)&smem[0];
    const unsigned lds_sub  = lds_base +
        (unsigned)((tw * TILE_ELEMS + hh * 64 * LDS_STRIDE) * 2);
#else
    const int lrow = tid >> 1;
    const int lch  = (tid & 1) * 2;
    const __bf16* xg = xh  + ((size_t)t * B + mb + lrow) * C;
    const __bf16* tg = th  + ((size_t)t * B + mb + lrow) * C;
    const __bf16* sg = wsh + ((size_t)t * D + nb + lrow) * C;
    const __bf16* dg = wdh + ((size_t)t * D + nb + lrow) * C;
#endif

    v8f acc[2][4];
#pragma unroll
    for (int i = 0; i < 2; ++i)
#pragma unroll
        for (int j = 0; j < 4; ++j)
            acc[i][j] = (v8f){0.f, 0.f, 0.f, 0.f, 0.f, 0.f, 0.f, 0.f};

    union Frag { v16bf v; uint4 q[2]; };

#if USE_TDM
    // prologue: stage K-step 0 into buffer 0
    tdm_load_subtile(lds_sub, gsub);
#endif

    for (int k = 0; k < NK; ++k) {
        const int buf = k & 1;

#if USE_TDM
        if (k + 1 < NK) {
            // previous compute on buffer (k+1)&1 finished at loop-tail barrier
            tdm_load_subtile(lds_sub + (unsigned)(((k + 1) & 1) * BUF_ELEMS * 2),
                             gsub + (unsigned long long)((k + 1) * BK) * 2ull);
            __builtin_amdgcn_s_wait_tensorcnt(1);   // my buf-k subtile done
        } else {
            __builtin_amdgcn_s_wait_tensorcnt(0);
        }
#else
        {
            const int kc = k * BK;
            __bf16* bp = &smem[buf * BUF_ELEMS];
            const uint4* s0 = reinterpret_cast<const uint4*>(xg + kc);
            const uint4* s1 = reinterpret_cast<const uint4*>(tg + kc);
            const uint4* s2 = reinterpret_cast<const uint4*>(sg + kc);
            const uint4* s3 = reinterpret_cast<const uint4*>(dg + kc);
            uint4* d0 = reinterpret_cast<uint4*>(&bp[0 * TILE_ELEMS + lrow * LDS_STRIDE]);
            uint4* d1 = reinterpret_cast<uint4*>(&bp[1 * TILE_ELEMS + lrow * LDS_STRIDE]);
            uint4* d2 = reinterpret_cast<uint4*>(&bp[2 * TILE_ELEMS + lrow * LDS_STRIDE]);
            uint4* d3 = reinterpret_cast<uint4*>(&bp[3 * TILE_ELEMS + lrow * LDS_STRIDE]);
            d0[lch] = s0[lch]; d0[lch + 1] = s0[lch + 1];
            d1[lch] = s1[lch]; d1[lch + 1] = s1[lch + 1];
            d2[lch] = s2[lch]; d2[lch + 1] = s2[lch + 1];
            d3[lch] = s3[lch]; d3[lch + 1] = s3[lch + 1];
        }
#endif
        __syncthreads();   // all 8 subtiles of buf are resident

        const __bf16* Axp = &smem[buf * BUF_ELEMS + 0 * TILE_ELEMS];
        const __bf16* Atp = &smem[buf * BUF_ELEMS + 1 * TILE_ELEMS];
        const __bf16* Bsp = &smem[buf * BUF_ELEMS + 2 * TILE_ELEMS];
        const __bf16* Bdp = &smem[buf * BUF_ELEMS + 3 * TILE_ELEMS];

        // ---- fragments per ISA 16-bit A/B lane layout ----
        Frag ax[2], atr[2], bs[4], bd[4];
#pragma unroll
        for (int mt = 0; mt < 2; ++mt) {
            int r = wm + mt * 16 + lo;
            ax[mt].q[0]  = *reinterpret_cast<const uint4*>(&Axp[r * LDS_STRIDE + hi * 8]);
            ax[mt].q[1]  = *reinterpret_cast<const uint4*>(&Axp[r * LDS_STRIDE + 16 + hi * 8]);
            atr[mt].q[0] = *reinterpret_cast<const uint4*>(&Atp[r * LDS_STRIDE + hi * 8]);
            atr[mt].q[1] = *reinterpret_cast<const uint4*>(&Atp[r * LDS_STRIDE + 16 + hi * 8]);
        }
#pragma unroll
        for (int nt = 0; nt < 4; ++nt) {
            int r = wn + nt * 16 + lo;
            bs[nt].q[0] = *reinterpret_cast<const uint4*>(&Bsp[r * LDS_STRIDE + hi * 8]);
            bs[nt].q[1] = *reinterpret_cast<const uint4*>(&Bsp[r * LDS_STRIDE + 16 + hi * 8]);
            bd[nt].q[0] = *reinterpret_cast<const uint4*>(&Bdp[r * LDS_STRIDE + hi * 8]);
            bd[nt].q[1] = *reinterpret_cast<const uint4*>(&Bdp[r * LDS_STRIDE + 16 + hi * 8]);
        }

        // ---- 16 WMMAs: x·Ws^T and trend·(Wt-Ws)^T, shared accumulators ----
#pragma unroll
        for (int mt = 0; mt < 2; ++mt)
#pragma unroll
            for (int nt = 0; nt < 4; ++nt) {
                acc[mt][nt] = __builtin_amdgcn_wmma_f32_16x16x32_bf16(
                    false, ax[mt].v, false, bs[nt].v, (short)0, acc[mt][nt], false, false);
                acc[mt][nt] = __builtin_amdgcn_wmma_f32_16x16x32_bf16(
                    false, atr[mt].v, false, bd[nt].v, (short)0, acc[mt][nt], false, false);
            }
        __syncthreads();   // buf free for overwrite next iteration
    }

    // ---- epilogue: + (b_s + b_t), store fp32 out[b][t][d] ----
#pragma unroll
    for (int nt = 0; nt < 4; ++nt) {
        int col = nb + wn + nt * 16 + lo;
        float bias = bsv[t * D + col] + btv[t * D + col];
#pragma unroll
        for (int mt = 0; mt < 2; ++mt) {
            int mrow = mb + wm + mt * 16 + hi * 8;   // C/D layout: M = r + 8*hi
#pragma unroll
            for (int r = 0; r < 8; ++r) {
                out[((size_t)(mrow + r) * T + t) * D + col] = acc[mt][nt][r] + bias;
            }
        }
    }
}

// ---------------------------------------------------------------------------
extern "C" void kernel_launch(void* const* d_in, const int* in_sizes, int n_in,
                              void* d_out, int out_size, void* d_ws, size_t ws_size,
                              hipStream_t stream)
{
    const float* x  = (const float*)d_in[0];
    const float* Ws = (const float*)d_in[1];
    const float* bs = (const float*)d_in[2];
    const float* Wt = (const float*)d_in[3];
    const float* bt = (const float*)d_in[4];
    float* out = (float*)d_out;

    const int B = in_sizes[0] / (T * C);   // 4096

    // workspace carve-out (bf16): xh, th : [T][B][C]; wsh, wdh : [T][D][C]
    const size_t nx = (size_t)T * B * C;
    const size_t nw = (size_t)T * D * C;
    char* ws = (char*)d_ws;
    __bf16* xh  = (__bf16*)ws;
    __bf16* th  = (__bf16*)(ws + nx * 2);
    __bf16* wsh = (__bf16*)(ws + 2 * nx * 2);
    __bf16* wdh = (__bf16*)(ws + 2 * nx * 2 + nw * 2);

    {
        int total = B * (C / 4);
        prep_x_kernel<<<(total + 255) / 256, 256, 0, stream>>>(x, xh, th, B);
    }
    {
        int n4 = (int)(nw / 4);
        prep_w_kernel<<<(n4 + 255) / 256, 256, 0, stream>>>(Ws, Wt, wsh, wdh, n4);
    }
    {
        dim3 grid(B / BM, D / BN, T);
        dlinear_gemm_kernel<<<grid, 256, 0, stream>>>(xh, th, wsh, wdh, bs, bt, out, B);
    }
}
